// GraphAttentionEncoder_27144193311548
// MI455X (gfx1250) — matrix operand order repaired and checked by reading, not compiled, not run on hardware
//
#include <hip/hip_runtime.h>
#include <math.h>

#define DIM 128
#define NH 4
#define CH 32
#define LRELU 0.2f
#define LN_EPS 1e-5f

typedef __attribute__((ext_vector_type(16))) _Float16 v16h;
typedef __attribute__((ext_vector_type(8)))  float    v8f;

__device__ __forceinline__ float warpSum(float v) {
#pragma unroll
    for (int o = 16; o > 0; o >>= 1) v += __shfl_xor(v, o, 32);
    return v;
}

// order-consistent float atomic max (init memory to -inf)
__device__ __forceinline__ void atomicMaxF(float* a, float v) {
    if (v >= 0.0f) atomicMax((int*)a, __float_as_int(v));
    else           atomicMin((unsigned int*)a, __float_as_uint(v));
}

// ---------------- GEMM: h = x @ W + b
// Block = 256 threads (8 waves). blockIdx.y selects the 16-wide N tile; the
// f16 weight tile (128x16 -> stored [n][k], 4KB) is staged in LDS once and
// reused by all 8 waves; each wave computes one 16x16 output tile with 4
// fully-unrolled v_wmma_f32_16x16x32_f16 ops.
__global__ void gat_gemm_wmma(const float* __restrict__ x,
                              const float* __restrict__ W1, const float* __restrict__ b1,
                              const float* __restrict__ W2, const float* __restrict__ b2,
                              float* __restrict__ h1, float* __restrict__ h2, int nodes) {
    __shared__ _Float16 Bs[16 * DIM];   // Bs[n][k]  (n = col within tile)

    const int layer = blockIdx.z;
    const float* W    = layer ? W2 : W1;
    const float* bias = layer ? b2 : b1;
    float* out        = layer ? h2 : h1;

    const int colBase = blockIdx.y * 16;

    // cooperative fill: 2048 f16 elements, 8 per thread
#pragma unroll
    for (int i = 0; i < 8; i++) {
        int idx = threadIdx.x * 8 + i;
        int n = idx >> 7, k = idx & 127;
        Bs[n * DIM + k] = (_Float16)W[k * DIM + colBase + n];
    }
    __syncthreads();

    const int wave = threadIdx.x >> 5;
    const int lane = threadIdx.x & 31;
    const int half = lane >> 4;         // 0: lanes 0-15, 1: lanes 16-31
    const int lm   = lane & 15;

    const int mTiles = (nodes + 15) / 16;
    const int tM = blockIdx.x * 8 + wave;
    if (tM >= mTiles) return;

    const int row = tM * 16 + lm;
    const int rcl = row < nodes ? row : (nodes - 1);
    const int col = colBase + lm;
    const float* xr = x + (size_t)rcl * DIM;

    v8f acc = {};
#pragma unroll
    for (int k0 = 0; k0 < DIM; k0 += 32) {
        // A 16x32 (MxK): lane half 0 holds K = k0+0..15, half 1 holds k0+16..31
        float4 a0 = *(const float4*)(xr + k0 + half * 16);
        float4 a1 = *(const float4*)(xr + k0 + half * 16 + 4);
        float4 a2 = *(const float4*)(xr + k0 + half * 16 + 8);
        float4 a3 = *(const float4*)(xr + k0 + half * 16 + 12);
        v16h a;
        a[0]=(_Float16)a0.x; a[1]=(_Float16)a0.y; a[2]=(_Float16)a0.z; a[3]=(_Float16)a0.w;
        a[4]=(_Float16)a1.x; a[5]=(_Float16)a1.y; a[6]=(_Float16)a1.z; a[7]=(_Float16)a1.w;
        a[8]=(_Float16)a2.x; a[9]=(_Float16)a2.y; a[10]=(_Float16)a2.z; a[11]=(_Float16)a2.w;
        a[12]=(_Float16)a3.x; a[13]=(_Float16)a3.y; a[14]=(_Float16)a3.z; a[15]=(_Float16)a3.w;
        // B 32x16 (KxN): lane half 0 holds K = k0+0..15, half 1 holds k0+16..31
        v16h b = *(const v16h*)(Bs + lm * DIM + k0 + half * 16);
        acc = __builtin_amdgcn_wmma_f32_16x16x32_f16(false, a, false, b,
                                                     (short)0, acc, false, false);
    }

    const float bc = bias[col];
    const bool fullTile = (tM * 16 + 16) <= nodes;
    if (fullTile) {
#pragma unroll
        for (int r = 0; r < 8; r++) {
            int m = tM * 16 + r + half * 8;   // C/D: VGPR r -> M=r (lo), M=r+8 (hi)
            out[(size_t)m * DIM + col] = acc[r] + bc;
        }
    } else {
#pragma unroll
        for (int r = 0; r < 8; r++) {
            int m = tM * 16 + r + half * 8;
            if (m < nodes) out[(size_t)m * DIM + col] = acc[r] + bc;
        }
    }
}

// ---------------- per-node attention logits e_src/e_dst for both layers
__global__ void gat_logits(const float* __restrict__ h1, const float* __restrict__ h2,
                           const float* __restrict__ as1, const float* __restrict__ ad1,
                           const float* __restrict__ as2, const float* __restrict__ ad2,
                           float* es1, float* ed1, float* es2, float* ed2, int nodes) {
    int t = blockIdx.x * blockDim.x + threadIdx.x;
    if (t >= nodes * NH) return;
    int n = t >> 2, hh = t & 3;
    const float* r1 = h1 + (size_t)n * DIM + hh * CH;
    const float* r2 = h2 + (size_t)n * DIM + hh * CH;
    float s1 = 0.f, d1 = 0.f, s2 = 0.f, d2 = 0.f;
#pragma unroll 8
    for (int c = 0; c < CH; c++) {
        s1 += r1[c] * as1[hh * CH + c];
        d1 += r1[c] * ad1[hh * CH + c];
        s2 += r2[c] * as2[hh * CH + c];
        d2 += r2[c] * ad2[hh * CH + c];
    }
    es1[t] = s1; ed1[t] = d1; es2[t] = s2; ed2[t] = d2;
}

// ---------------- init accumulators (ws is poisoned; must re-init every call)
__global__ void gat_init(float* m1, float* m2, float* s1, float* s2,
                         float* agg1, float* agg2, int nodes) {
    int t = blockIdx.x * blockDim.x + threadIdx.x;
    if (t < nodes * NH) {
        float ninf = __uint_as_float(0xFF800000u);
        m1[t] = ninf; m2[t] = ninf; s1[t] = 0.f; s2[t] = 0.f;
    }
    if (t < nodes * DIM) { agg1[t] = 0.f; agg2[t] = 0.f; }
}

// ---------------- edge pass 1: segment max of leaky-relu logits
__global__ void gat_edge_max(const int* __restrict__ ei, int E, int nodes,
                             const float* __restrict__ es1, const float* __restrict__ ed1,
                             const float* __restrict__ es2, const float* __restrict__ ed2,
                             float* m1, float* m2) {
    int e = blockIdx.x * blockDim.x + threadIdx.x;
    int ET = E + nodes;
    if (e >= ET) return;
    int src = (e < E) ? ei[e]     : (e - E);
    int dst = (e < E) ? ei[E + e] : (e - E);
#pragma unroll
    for (int hh = 0; hh < NH; hh++) {
        float v1 = es1[src * NH + hh] + ed1[dst * NH + hh];
        v1 = v1 > 0.f ? v1 : LRELU * v1;
        atomicMaxF(&m1[dst * NH + hh], v1);
        float v2 = es2[src * NH + hh] + ed2[dst * NH + hh];
        v2 = v2 > 0.f ? v2 : LRELU * v2;
        atomicMaxF(&m2[dst * NH + hh], v2);
    }
}

// ---------------- edge pass 2: p = exp(e - m[dst]); s[dst]+=p; agg[dst]+=p*h[src]
__global__ void gat_edge_agg(const int* __restrict__ ei, int E, int nodes,
                             const float* __restrict__ es1, const float* __restrict__ ed1,
                             const float* __restrict__ es2, const float* __restrict__ ed2,
                             const float* __restrict__ m1, const float* __restrict__ m2,
                             const float* __restrict__ h1, const float* __restrict__ h2,
                             float* s1, float* s2, float* agg1, float* agg2) {
    int e = blockIdx.x * blockDim.x + threadIdx.x;
    int ET = E + nodes;
    if (e >= ET) return;
    int src = (e < E) ? ei[e]     : (e - E);
    int dst = (e < E) ? ei[E + e] : (e - E);

    const float* r1 = h1 + (size_t)src * DIM;
    const float* r2 = h2 + (size_t)src * DIM;
    // hint the gathered source rows into cache ahead of the logit math
    __builtin_prefetch(r1, 0, 3);
    __builtin_prefetch(r2, 0, 3);

    float p1[NH], p2[NH];
#pragma unroll
    for (int hh = 0; hh < NH; hh++) {
        float v1 = es1[src * NH + hh] + ed1[dst * NH + hh];
        v1 = v1 > 0.f ? v1 : LRELU * v1;
        p1[hh] = __expf(v1 - m1[dst * NH + hh]);
        atomicAdd(&s1[dst * NH + hh], p1[hh]);
        float v2 = es2[src * NH + hh] + ed2[dst * NH + hh];
        v2 = v2 > 0.f ? v2 : LRELU * v2;
        p2[hh] = __expf(v2 - m2[dst * NH + hh]);
        atomicAdd(&s2[dst * NH + hh], p2[hh]);
    }
    float* a1 = agg1 + (size_t)dst * DIM;
    float* a2 = agg2 + (size_t)dst * DIM;
#pragma unroll 2
    for (int d = 0; d < DIM; d += 4) {
        float w1 = p1[d >> 5], w2 = p2[d >> 5];
        float4 v1 = *(const float4*)(r1 + d);
        float4 v2 = *(const float4*)(r2 + d);
        atomicAdd(&a1[d + 0], w1 * v1.x);
        atomicAdd(&a1[d + 1], w1 * v1.y);
        atomicAdd(&a1[d + 2], w1 * v1.z);
        atomicAdd(&a1[d + 3], w1 * v1.w);
        atomicAdd(&a2[d + 0], w2 * v2.x);
        atomicAdd(&a2[d + 1], w2 * v2.y);
        atomicAdd(&a2[d + 2], w2 * v2.z);
        atomicAdd(&a2[d + 3], w2 * v2.w);
    }
}

// ---------------- per-node epilogue: normalize, gate softmax, residual, LayerNorm
__global__ void gat_finalize(const float* __restrict__ x,
                             const float* __restrict__ agg1, const float* __restrict__ s1,
                             const float* __restrict__ b1,
                             const float* __restrict__ agg2, const float* __restrict__ s2,
                             const float* __restrict__ b2,
                             const float* __restrict__ gW, const float* __restrict__ gb,
                             const float* __restrict__ gamma, const float* __restrict__ beta,
                             float* __restrict__ out, int nodes) {
    int node = blockIdx.x * (blockDim.x >> 5) + (threadIdx.x >> 5);
    int lane = threadIdx.x & 31;
    if (node >= nodes) return;
    int d0 = lane * 4;
    int hh = lane >> 3;                       // head of this lane's 4 channels

    float4 a1 = *(const float4*)(agg1 + (size_t)node * DIM + d0);
    float4 a2 = *(const float4*)(agg2 + (size_t)node * DIM + d0);
    float inv1 = 1.0f / s1[node * NH + hh];
    float inv2 = 1.0f / s2[node * NH + hh];
    float o1[4] = { a1.x * inv1 + b1[d0], a1.y * inv1 + b1[d0 + 1],
                    a1.z * inv1 + b1[d0 + 2], a1.w * inv1 + b1[d0 + 3] };
    float o2[4] = { a2.x * inv2 + b2[d0], a2.y * inv2 + b2[d0 + 1],
                    a2.z * inv2 + b2[d0 + 2], a2.w * inv2 + b2[d0 + 3] };

    // gate logits: concat(out1,out2) @ gate_W  (gate_W is [256,2] row-major)
    float l0 = 0.f, l1 = 0.f;
#pragma unroll
    for (int j = 0; j < 4; j++) {
        int d = d0 + j;
        l0 += o1[j] * gW[d * 2 + 0] + o2[j] * gW[(DIM + d) * 2 + 0];
        l1 += o1[j] * gW[d * 2 + 1] + o2[j] * gW[(DIM + d) * 2 + 1];
    }
    l0 = warpSum(l0) + gb[0];
    l1 = warpSum(l1) + gb[1];
    float gm = fmaxf(l0, l1);
    float e0 = __expf(l0 - gm), e1 = __expf(l1 - gm);
    float g0 = e0 / (e0 + e1), g1 = e1 / (e0 + e1);

    float4 xv = *(const float4*)(x + (size_t)node * DIM + d0);
    float y[4];
    const float* xs = (const float*)&xv;
#pragma unroll
    for (int j = 0; j < 4; j++) y[j] = xs[j] + g0 * o1[j] + g1 * o2[j];

    float sum = 0.f, sq = 0.f;
#pragma unroll
    for (int j = 0; j < 4; j++) { sum += y[j]; sq += y[j] * y[j]; }
    sum = warpSum(sum); sq = warpSum(sq);
    float mean = sum * (1.0f / DIM);
    float var  = sq  * (1.0f / DIM) - mean * mean;
    float inv  = rsqrtf(var + LN_EPS);
#pragma unroll
    for (int j = 0; j < 4; j++) {
        int d = d0 + j;
        out[(size_t)node * DIM + d] = (y[j] - mean) * inv * gamma[d] + beta[d];
    }
}

extern "C" void kernel_launch(void* const* d_in, const int* in_sizes, int n_in,
                              void* d_out, int out_size, void* d_ws, size_t ws_size,
                              hipStream_t stream) {
    const float* x    = (const float*)d_in[0];
    const int*   ei   = (const int*)d_in[1];
    const float* W1   = (const float*)d_in[2];
    const float* b1   = (const float*)d_in[3];
    const float* as1  = (const float*)d_in[4];
    const float* ad1  = (const float*)d_in[5];
    const float* W2   = (const float*)d_in[6];
    const float* b2   = (const float*)d_in[7];
    const float* as2  = (const float*)d_in[8];
    const float* ad2  = (const float*)d_in[9];
    const float* gW   = (const float*)d_in[10];
    const float* gb   = (const float*)d_in[11];
    const float* gam  = (const float*)d_in[12];
    const float* bet  = (const float*)d_in[13];
    float* out = (float*)d_out;

    const int nodes = in_sizes[0] / DIM;
    const int E     = in_sizes[1] / 2;
    const int ET    = E + nodes;

    // workspace layout (floats)
    float* ws  = (float*)d_ws;
    size_t nn  = (size_t)nodes;
    float* h1   = ws;
    float* h2   = h1  + nn * DIM;
    float* es1  = h2  + nn * DIM;
    float* ed1  = es1 + nn * NH;
    float* es2  = ed1 + nn * NH;
    float* ed2  = es2 + nn * NH;
    float* m1   = ed2 + nn * NH;
    float* m2   = m1  + nn * NH;
    float* s1   = m2  + nn * NH;
    float* s2   = s1  + nn * NH;
    float* agg1 = s2  + nn * NH;
    float* agg2 = agg1 + nn * DIM;

    const int mTiles = (nodes + 15) / 16;
    dim3 ggrid((mTiles + 7) / 8, DIM / 16, 2);
    gat_gemm_wmma<<<ggrid, 256, 0, stream>>>(x, W1, b1, W2, b2, h1, h2, nodes);

    gat_logits<<<(nodes * NH + 255) / 256, 256, 0, stream>>>(
        h1, h2, as1, ad1, as2, ad2, es1, ed1, es2, ed2, nodes);

    gat_init<<<(nodes * DIM + 255) / 256, 256, 0, stream>>>(m1, m2, s1, s2, agg1, agg2, nodes);

    gat_edge_max<<<(ET + 255) / 256, 256, 0, stream>>>(ei, E, nodes, es1, ed1, es2, ed2, m1, m2);

    gat_edge_agg<<<(ET + 255) / 256, 256, 0, stream>>>(
        ei, E, nodes, es1, ed1, es2, ed2, m1, m2, h1, h2, s1, s2, agg1, agg2);

    gat_finalize<<<(nodes + 3) / 4, 128, 0, stream>>>(
        x, agg1, s1, b1, agg2, s2, b2, gW, gb, gam, bet, out, nodes);
}
